// SetAbstraction_14654428414839
// MI455X (gfx1250) — compile-verified, matrix operand order checked by reading, not run on hardware
//
#include <hip/hip_runtime.h>
#include <hip/hip_bf16.h>

#define BATCH 16
#define NPTS  4096
#define NCENT 1024
#define KNB   32
#define CFEAT 64
#define DIN   67
#define KPAD  96
#define H1C   64
#define H2C   64
#define HOUTC 128
#define LNEPS 1e-5f

typedef __attribute__((ext_vector_type(16))) _Float16 v16h;
typedef __attribute__((ext_vector_type(2)))  __fp16   v2hp;  // cvt_pkrtz result
typedef __attribute__((ext_vector_type(8)))  float    v8f;

// K-index inside a 32-wide k-step for operand half i (ISA wave32 f16 layout):
// halves 0..7 -> K = oct*8 + 0..7 ; halves 8..15 -> K = oct*8 + 16..23
__host__ __device__ __forceinline__ int kmap(int i) { return (i < 8) ? i : i + 8; }

// ---------------------------------------------------------------------------
// Kernel 1: farthest point sampling, one block per batch. Writes centroids
// (B, M, 3) directly into the first region of d_out.
// ---------------------------------------------------------------------------
__global__ void __launch_bounds__(256) fps_kernel(const float* __restrict__ xyz,
                                                  float* __restrict__ cent) {
  const int b = blockIdx.x;
  const int t = threadIdx.x;
  const float* pts = xyz + (size_t)b * NPTS * 3;
  __shared__ float sval[256];
  __shared__ int   sidx[256];
  __shared__ float scur[3];

  float px[16], py[16], pz[16], dmin[16];
#pragma unroll
  for (int i = 0; i < 16; ++i) {
    int n = t + i * 256;
    px[i] = pts[n * 3 + 0];
    py[i] = pts[n * 3 + 1];
    pz[i] = pts[n * 3 + 2];
  }
  float cx = pts[0], cy = pts[1], cz = pts[2];
  if (t == 0) {
    cent[(size_t)(b * NCENT) * 3 + 0] = cx;
    cent[(size_t)(b * NCENT) * 3 + 1] = cy;
    cent[(size_t)(b * NCENT) * 3 + 2] = cz;
  }
#pragma unroll
  for (int i = 0; i < 16; ++i) {
    float dx = px[i] - cx, dy = py[i] - cy, dz = pz[i] - cz;
    dmin[i] = dx * dx + dy * dy + dz * dz;
  }
  for (int s = 1; s < NCENT; ++s) {
    float bv = -1.0f;
    int   bi = 0x7fffffff;
#pragma unroll
    for (int i = 0; i < 16; ++i) {
      int n = t + i * 256;
      if (dmin[i] > bv) { bv = dmin[i]; bi = n; }
    }
    sval[t] = bv;
    sidx[t] = bi;
    __syncthreads();
    for (int off = 128; off > 0; off >>= 1) {
      if (t < off) {
        float ov = sval[t + off];
        int   oi = sidx[t + off];
        if (ov > sval[t] || (ov == sval[t] && oi < sidx[t])) {
          sval[t] = ov;
          sidx[t] = oi;
        }
      }
      __syncthreads();
    }
    if (t == 0) {
      int nxt = sidx[0];
      float nx = pts[nxt * 3 + 0];
      float ny = pts[nxt * 3 + 1];
      float nz = pts[nxt * 3 + 2];
      scur[0] = nx; scur[1] = ny; scur[2] = nz;
      cent[(size_t)(b * NCENT + s) * 3 + 0] = nx;
      cent[(size_t)(b * NCENT + s) * 3 + 1] = ny;
      cent[(size_t)(b * NCENT + s) * 3 + 2] = nz;
    }
    __syncthreads();
    cx = scur[0]; cy = scur[1]; cz = scur[2];
#pragma unroll
    for (int i = 0; i < 16; ++i) {
      float dx = px[i] - cx, dy = py[i] - cy, dz = pz[i] - cz;
      float d = dx * dx + dy * dy + dz * dz;
      dmin[i] = fminf(dmin[i], d);
    }
  }
}

// ---------------------------------------------------------------------------
// Kernel 2: pre-swizzle Linear weights (out,in) f32 into WMMA B-fragment
// blobs: frag(nt,kt) = 32 lanes x 16 halves, lane-contiguous (32 B / lane),
// so each B fragment in the MLP kernel is one 32-byte v16h load.
//   half i of lane l in frag(nt,kt) = W[n = nt*16 + (l&15)]
//                                      [k = kt*32 + (l>>4)*8 + kmap(i)]
// ---------------------------------------------------------------------------
__global__ void __launch_bounds__(256) prep_weights(
    const float* __restrict__ W1, const float* __restrict__ W2,
    const float* __restrict__ Wout,
    _Float16* __restrict__ w1B, _Float16* __restrict__ w2B,
    _Float16* __restrict__ woB) {
  int o = blockIdx.x * blockDim.x + threadIdx.x;
  // layer 1: 4 nt x 3 kt fragments, K padded 67 -> 96
  if (o < 4 * 3 * 512) {
    int f = o >> 9, r = o & 511, lane = r >> 4, i = r & 15;
    int nt = f / 3, kt = f % 3;
    int col = nt * 16 + (lane & 15);
    int k = kt * 32 + (lane >> 4) * 8 + kmap(i);
    w1B[o] = (k < DIN) ? (_Float16)W1[col * DIN + k] : (_Float16)0.0f;
  }
  // layer 2: 4 nt x 2 kt
  if (o < 4 * 2 * 512) {
    int f = o >> 9, r = o & 511, lane = r >> 4, i = r & 15;
    int nt = f / 2, kt = f % 2;
    int col = nt * 16 + (lane & 15);
    int k = kt * 32 + (lane >> 4) * 8 + kmap(i);
    w2B[o] = (_Float16)W2[col * H1C + k];
  }
  // layer 3: 8 nt x 2 kt
  if (o < 8 * 2 * 512) {
    int f = o >> 9, r = o & 511, lane = r >> 4, i = r & 15;
    int nt = f / 2, kt = f % 2;
    int col = nt * 16 + (lane & 15);
    int k = kt * 32 + (lane >> 4) * 8 + kmap(i);
    woB[o] = (_Float16)Wout[col * H2C + k];
  }
}

// ---------------------------------------------------------------------------
// Kernel 3: KNN, one block per centroid. d2 table in LDS, 32 rounds of
// argmin with masking (matches top_k tie behavior: lowest index wins).
// ---------------------------------------------------------------------------
__global__ void __launch_bounds__(256) knn_kernel(const float* __restrict__ xyz,
                                                  const float* __restrict__ cent,
                                                  int* __restrict__ gidx) {
  const int cid = blockIdx.x;  // b*NCENT + m
  const int b = cid / NCENT;
  const int t = threadIdx.x;
  const float* pts = xyz + (size_t)b * NPTS * 3;
  const float cx = cent[(size_t)cid * 3 + 0];
  const float cy = cent[(size_t)cid * 3 + 1];
  const float cz = cent[(size_t)cid * 3 + 2];
  __shared__ float d2s[NPTS];
  __shared__ float sval[256];
  __shared__ int   sidx[256];

#pragma unroll
  for (int i = 0; i < 16; ++i) {
    int n = t + i * 256;
    float dx = pts[n * 3 + 0] - cx;
    float dy = pts[n * 3 + 1] - cy;
    float dz = pts[n * 3 + 2] - cz;
    d2s[n] = dx * dx + dy * dy + dz * dz;
  }
  __syncthreads();
  for (int s = 0; s < KNB; ++s) {
    float bv = 3.4e38f;
    int   bi = 0x7fffffff;
#pragma unroll
    for (int i = 0; i < 16; ++i) {
      int n = t + i * 256;
      float v = d2s[n];
      if (v < bv || (v == bv && n < bi)) { bv = v; bi = n; }
    }
    sval[t] = bv;
    sidx[t] = bi;
    __syncthreads();
    for (int off = 128; off > 0; off >>= 1) {
      if (t < off) {
        float ov = sval[t + off];
        int   oi = sidx[t + off];
        if (ov < sval[t] || (ov == sval[t] && oi < sidx[t])) {
          sval[t] = ov;
          sidx[t] = oi;
        }
      }
      __syncthreads();
    }
    if (t == 0) {
      gidx[(size_t)cid * KNB + s] = sidx[0];
      d2s[sidx[0]] = 3.4e38f;
    }
    __syncthreads();
  }
}

// ---------------------------------------------------------------------------
// Kernel 4: fused gather -> (Linear,LN,ReLU)x2 -> Linear -> maxpool.
// One wave per centroid. Activations live in LDS already swizzled into
// A-fragment chunk order; every WMMA operand is a single 32-B vector access.
// ---------------------------------------------------------------------------
// pack 16 f32 (chunk order) -> v16h using v_cvt_pkrtz pairs
__device__ __forceinline__ v16h pack_chunk(const float* x, int kbase) {
  union { v16h v; v2hp p[8]; } u;
#pragma unroll
  for (int j = 0; j < 8; ++j) {
    int k = kbase + kmap(2 * j);
    u.p[j] = __builtin_amdgcn_cvt_pkrtz(x[k], x[k + 1]);
  }
  return u.v;
}

// LayerNorm + ReLU of one 64-wide row; writes 4 swizzled 32-B chunks.
__device__ __forceinline__ void ln_relu_row(const float* __restrict__ h,
                                            const float* __restrict__ g,
                                            const float* __restrict__ be,
                                            _Float16* __restrict__ orow) {
  float v[64];
  const float4* h4 = (const float4*)h;
#pragma unroll
  for (int q = 0; q < 16; ++q) {
    float4 tq = h4[q];
    v[4 * q + 0] = tq.x; v[4 * q + 1] = tq.y;
    v[4 * q + 2] = tq.z; v[4 * q + 3] = tq.w;
  }
  float mu = 0.f;
#pragma unroll
  for (int k = 0; k < 64; ++k) mu += v[k];
  mu *= (1.0f / 64.0f);
  float var = 0.f;
#pragma unroll
  for (int k = 0; k < 64; ++k) {
    float d = v[k] - mu;
    var += d * d;
  }
  var *= (1.0f / 64.0f);
  float rs = rsqrtf(var + LNEPS);
  const float4* g4 = (const float4*)g;
  const float4* be4 = (const float4*)be;
  float y[64];
#pragma unroll
  for (int q = 0; q < 16; ++q) {
    float4 gq = g4[q], bq = be4[q];
    y[4 * q + 0] = fmaxf((v[4 * q + 0] - mu) * rs * gq.x + bq.x, 0.f);
    y[4 * q + 1] = fmaxf((v[4 * q + 1] - mu) * rs * gq.y + bq.y, 0.f);
    y[4 * q + 2] = fmaxf((v[4 * q + 2] - mu) * rs * gq.z + bq.z, 0.f);
    y[4 * q + 3] = fmaxf((v[4 * q + 3] - mu) * rs * gq.w + bq.w, 0.f);
  }
#pragma unroll
  for (int c = 0; c < 4; ++c)
    *(v16h*)&orow[c * 16] = pack_chunk(y, (c >> 1) * 32 + (c & 1) * 8);
}

__global__ void __launch_bounds__(32) mlp_kernel(
    const float* __restrict__ xyz, const float* __restrict__ feat,
    const float* __restrict__ cent, const int* __restrict__ gidx,
    const _Float16* __restrict__ w1B, const float* __restrict__ b1,
    const float* __restrict__ g1, const float* __restrict__ be1,
    const _Float16* __restrict__ w2B, const float* __restrict__ b2,
    const float* __restrict__ g2, const float* __restrict__ be2,
    const _Float16* __restrict__ woB, const float* __restrict__ bout,
    float* __restrict__ out) {
  const int cid = blockIdx.x;  // b*NCENT + m
  const int b = cid / NCENT;
  const int lane = threadIdx.x;

  // rows stored as 32-B chunks in A-fragment order; +pad chunk for banks
  __shared__ __align__(32) _Float16 xs[32 * 7 * 16];  // layer-1 A (6 chunks)
  __shared__ __align__(16) float    hf[32 * 64];      // f32 pre-LN
  __shared__ __align__(32) _Float16 hs[32 * 5 * 16];  // layers-2/3 A (4 chunks)

  const v16h* w1f = (const v16h*)w1B;
  const v16h* w2f = (const v16h*)w2B;
  const v16h* wof = (const v16h*)woB;

  // ---- gather: lane j builds row j = [xyz - centroid | features | 0-pad]
  {
    int n = gidx[(size_t)cid * KNB + lane];
    float x[KPAD];
    const float* p = xyz + ((size_t)b * NPTS + n) * 3;
    x[0] = p[0] - cent[(size_t)cid * 3 + 0];
    x[1] = p[1] - cent[(size_t)cid * 3 + 1];
    x[2] = p[2] - cent[(size_t)cid * 3 + 2];
    const float4* f4 = (const float4*)(feat + ((size_t)b * NPTS + n) * CFEAT);
#pragma unroll
    for (int q = 0; q < 16; ++q) {
      float4 tq = f4[q];
      x[3 + 4 * q + 0] = tq.x; x[3 + 4 * q + 1] = tq.y;
      x[3 + 4 * q + 2] = tq.z; x[3 + 4 * q + 3] = tq.w;
    }
#pragma unroll
    for (int k = DIN; k < KPAD; ++k) x[k] = 0.f;
#pragma unroll
    for (int c = 0; c < 6; ++c)
      *(v16h*)&xs[(lane * 7 + c) * 16] =
          pack_chunk(x, (c >> 1) * 32 + (c & 1) * 8);
  }
  __syncthreads();

  const int colw = lane & 15;  // C/D column within tile; A row within tile
  const int half = lane >> 4;  // K-octet selector

  // ================= layer 1: (32 x 96) @ (96 x 64) =================
  v16h A1[2][3];
#pragma unroll
  for (int mt = 0; mt < 2; ++mt)
#pragma unroll
    for (int kt = 0; kt < 3; ++kt)
      A1[mt][kt] =
          *(const v16h*)&xs[((mt * 16 + colw) * 7 + (kt * 2 + half)) * 16];
#pragma unroll
  for (int nt = 0; nt < 4; ++nt) {
    int col = nt * 16 + colw;
    float bv = b1[col];
    v8f c0, c1;
#pragma unroll
    for (int r = 0; r < 8; ++r) { c0[r] = bv; c1[r] = bv; }
#pragma unroll
    for (int kt = 0; kt < 3; ++kt) {
      v16h Bf = w1f[(nt * 3 + kt) * 32 + lane];
      c0 = __builtin_amdgcn_wmma_f32_16x16x32_f16(false, A1[0][kt], false, Bf,
                                                  (short)0, c0, false, false);
      c1 = __builtin_amdgcn_wmma_f32_16x16x32_f16(false, A1[1][kt], false, Bf,
                                                  (short)0, c1, false, false);
    }
#pragma unroll
    for (int r = 0; r < 8; ++r) {
      hf[(half * 8 + r) * 64 + col] = c0[r];
      hf[(16 + half * 8 + r) * 64 + col] = c1[r];
    }
  }
  __syncthreads();
  ln_relu_row(&hf[lane * 64], g1, be1, &hs[lane * 5 * 16]);
  __syncthreads();

  // ================= layer 2: (32 x 64) @ (64 x 64) =================
  v16h A2[2][2];
#pragma unroll
  for (int mt = 0; mt < 2; ++mt)
#pragma unroll
    for (int kt = 0; kt < 2; ++kt)
      A2[mt][kt] =
          *(const v16h*)&hs[((mt * 16 + colw) * 5 + (kt * 2 + half)) * 16];
#pragma unroll
  for (int nt = 0; nt < 4; ++nt) {
    int col = nt * 16 + colw;
    float bv = b2[col];
    v8f c0, c1;
#pragma unroll
    for (int r = 0; r < 8; ++r) { c0[r] = bv; c1[r] = bv; }
#pragma unroll
    for (int kt = 0; kt < 2; ++kt) {
      v16h Bf = w2f[(nt * 2 + kt) * 32 + lane];
      c0 = __builtin_amdgcn_wmma_f32_16x16x32_f16(false, A2[0][kt], false, Bf,
                                                  (short)0, c0, false, false);
      c1 = __builtin_amdgcn_wmma_f32_16x16x32_f16(false, A2[1][kt], false, Bf,
                                                  (short)0, c1, false, false);
    }
#pragma unroll
    for (int r = 0; r < 8; ++r) {
      hf[(half * 8 + r) * 64 + col] = c0[r];
      hf[(16 + half * 8 + r) * 64 + col] = c1[r];
    }
  }
  __syncthreads();
  ln_relu_row(&hf[lane * 64], g2, be2, &hs[lane * 5 * 16]);
  __syncthreads();

  // ===== layer 3: (32 x 64) @ (64 x 128) + maxpool over the 32 rows =====
  v16h A3[2][2];
#pragma unroll
  for (int mt = 0; mt < 2; ++mt)
#pragma unroll
    for (int kt = 0; kt < 2; ++kt)
      A3[mt][kt] =
          *(const v16h*)&hs[((mt * 16 + colw) * 5 + (kt * 2 + half)) * 16];
  float* o = out + (size_t)cid * HOUTC;
#pragma unroll
  for (int nt = 0; nt < 8; ++nt) {
    int col = nt * 16 + colw;
    float bv = bout[col];
    v8f c0, c1;
#pragma unroll
    for (int r = 0; r < 8; ++r) { c0[r] = bv; c1[r] = bv; }
#pragma unroll
    for (int kt = 0; kt < 2; ++kt) {
      v16h Bf = wof[(nt * 2 + kt) * 32 + lane];
      c0 = __builtin_amdgcn_wmma_f32_16x16x32_f16(false, A3[0][kt], false, Bf,
                                                  (short)0, c0, false, false);
      c1 = __builtin_amdgcn_wmma_f32_16x16x32_f16(false, A3[1][kt], false, Bf,
                                                  (short)0, c1, false, false);
    }
    // max over the 16 rows held by this lane (both M tiles) ...
    float m = c0[0];
#pragma unroll
    for (int r = 1; r < 8; ++r) m = fmaxf(m, c0[r]);
#pragma unroll
    for (int r = 0; r < 8; ++r) m = fmaxf(m, c1[r]);
    // ... then fold the other row-half (lane ^ 16 holds the same column)
    m = fmaxf(m, __shfl_xor(m, 16, 32));
    if (lane < 16) o[col] = m;
  }
}

// ---------------------------------------------------------------------------
extern "C" void kernel_launch(void* const* d_in, const int* in_sizes, int n_in,
                              void* d_out, int out_size, void* d_ws,
                              size_t ws_size, hipStream_t stream) {
  (void)in_sizes; (void)n_in; (void)out_size; (void)ws_size;
  const float* xyz  = (const float*)d_in[0];
  const float* feat = (const float*)d_in[1];
  const float* W1   = (const float*)d_in[2];
  const float* b1   = (const float*)d_in[3];
  const float* g1   = (const float*)d_in[4];
  const float* be1  = (const float*)d_in[5];
  const float* W2   = (const float*)d_in[6];
  const float* b2   = (const float*)d_in[7];
  const float* g2   = (const float*)d_in[8];
  const float* be2  = (const float*)d_in[9];
  const float* Wout = (const float*)d_in[10];
  const float* bout = (const float*)d_in[11];

  float* outp = (float*)d_out;
  float* cent = outp;                              // (B, M, 3)
  float* o    = outp + (size_t)BATCH * NCENT * 3;  // (B, M, 128)

  // workspace: [group indices | swizzled f16 weight fragments]
  int* gidx = (int*)d_ws;
  _Float16* w1B =
      (_Float16*)((char*)d_ws + (size_t)BATCH * NCENT * KNB * sizeof(int));
  _Float16* w2B = w1B + 4 * 3 * 512;  // 12 frags * 512 halves
  _Float16* woB = w2B + 4 * 2 * 512;  // 8  frags * 512 halves

  fps_kernel<<<BATCH, 256, 0, stream>>>(xyz, cent);
  prep_weights<<<(8 * 2 * 512 + 255) / 256, 256, 0, stream>>>(W1, W2, Wout,
                                                              w1B, w2B, woB);
  knn_kernel<<<BATCH * NCENT, 256, 0, stream>>>(xyz, cent, gidx);
  mlp_kernel<<<BATCH * NCENT, 32, 0, stream>>>(
      xyz, feat, cent, gidx, w1B, b1, g1, be1, w2B, b2, g2, be2, woB, bout, o);
}
